// MoEConnectionProcessor_68642167325227
// MI455X (gfx1250) — matrix operand order repaired and checked by reading, not compiled, not run on hardware
//
#include <hip/hip_runtime.h>

#define NCELLS 19683
#define KNBR   26
#define DDIM   64
#define TILES  ((NCELLS + 15) / 16)
#define WPB    4

typedef __attribute__((ext_vector_type(16))) __bf16 v16bf;
typedef __attribute__((ext_vector_type(8)))  float  v8f;
typedef __attribute__((ext_vector_type(2)))  float  v2f;
typedef __attribute__((ext_vector_type(16))) short  v16s;
typedef __attribute__((ext_vector_type(8)))  short  v8s;

union BF16Frag { v16s s; v16bf b; };

// per-wave LDS layout (units: unsigned short)
#define XA_OFF   0        // 16 x 136 bf16 A-operand (cols 0..63 dynamic, 64..127 mean)
#define XH_OFF   2176     // 16 x 136 bf16 intermediate h (also reused as f32 [16][32])
#define MEAN_OFF 4352     // 3 x 16 x 64 bf16 (mean_l, mean_f, mean_d)
#define WROW_OFF 7424     // 16 x 8 f32: w0,w1,w2,_,cnt0,cnt1,cnt2,_
#define WAVE_LDS 7680

// packed bf16 weight offsets in d_ws (shorts); all have Kin=128 => 4 K-blocks
#define OFF_WL  0
#define OFF_WF1 8192
#define OFF_WF2 24576
#define OFF_WC1 32768
#define OFF_WC2 49152
#define OFF_WG1 57344

__device__ __forceinline__ unsigned short f2bf(float f) {
  unsigned u = __float_as_uint(f);
  u += 0x7fffu + ((u >> 16) & 1u);           // round-to-nearest-even
  return (unsigned short)(u >> 16);
}

__device__ __forceinline__ float fast_tanh(float x) {
  x = fminf(fmaxf(x, -15.0f), 15.0f);
  float e = __expf(2.0f * x);
  return (e - 1.0f) / (e + 1.0f);
}

// A-matrix 16x32 bf16 fragment: lane<16 -> row=lane, K {0..15};
// lane>=16 -> row=lane-16, K {16..31}. Source: LDS row-major, stride 136.
__device__ __forceinline__ v16bf load_a_frag(const unsigned short* A, int lane, int kb) {
  int r = lane & 15;
  int kbase = kb * 32 + ((lane >> 4) << 4);
  const unsigned short* p = A + r * 136 + kbase;
  v8s lo = *(const v8s*)p;          // K kbase .. kbase+7
  v8s hi = *(const v8s*)(p + 8);    // K kbase+8 .. kbase+15
  BF16Frag u;
#pragma unroll
  for (int i = 0; i < 8; ++i) { u.s[i] = lo[i]; u.s[8 + i] = hi[i]; }
  return u.b;
}

// B-matrix 32x16 bf16 fragment, pre-packed in fragment order:
// element j of lane L = W[kb*32 + (L>=16?16:0) + j][nt*16 + (L&15)]
__device__ __forceinline__ v16bf load_b_frag(const unsigned short* W, int lane, int nt, int kb) {
  const unsigned short* p = W + (size_t)(((nt * 4) + kb) * 32 + lane) * 16;
  v8s lo = *(const v8s*)p;
  v8s hi = *(const v8s*)(p + 8);
  BF16Frag u;
#pragma unroll
  for (int i = 0; i < 8; ++i) { u.s[i] = lo[i]; u.s[8 + i] = hi[i]; }
  return u.b;
}

// [16 x 128] @ [128 x (NT*16)]; each 16x16 tile is consumed by `epi(t, acc)`
// immediately after its 4-step WMMA chain -> only ONE v8f accumulator live per tile.
template <int NT, typename Epi>
__device__ __forceinline__ void gemm128_fused(const unsigned short* A, const unsigned short* W,
                                              int lane, Epi epi) {
  v16bf a[4];
#pragma unroll
  for (int kb = 0; kb < 4; ++kb) { a[kb] = load_a_frag(A, lane, kb); }
#pragma unroll
  for (int t = 0; t < NT; ++t) {
    v8f c;
#pragma unroll
    for (int i = 0; i < 8; ++i) { c[i] = 0.0f; }
#pragma unroll
    for (int kb = 0; kb < 4; ++kb) {
      v16bf b = load_b_frag(W, lane, t, kb);
      c = __builtin_amdgcn_wmma_f32_16x16x32_bf16(false, a[kb], false, b, (short)0, c,
                                                  false, false);
    }
    epi(t, c);
  }
}

// copy one 16x64 bf16 mean buffer into XA columns 64..127 (16B chunks)
__device__ __forceinline__ void copy_mean(unsigned short* lds, int which, int lane) {
#pragma unroll
  for (int j = lane; j < 128; j += 32) {
    int r = j >> 3, ch = j & 7;
    uint4 v = *(const uint4*)(lds + MEAN_OFF + which * 1024 + r * 64 + ch * 8);
    *(uint4*)(lds + XA_OFF + r * 136 + 64 + ch * 8) = v;
  }
}

// fetch count-masked gate weight (index c = 0,1,2) for the 8 rows this lane touches
__device__ __forceinline__ void load_gate_w(const float* wrow, int lane, int c, float* w8) {
  int base8 = (lane >> 4) << 3;
#pragma unroll
  for (int v = 0; v < 8; ++v) {
    int m = v + base8;
    w8[v] = (wrow[m * 8 + 4 + c] > 0.0f) ? wrow[m * 8 + c] : 0.0f;
  }
}

__global__ __launch_bounds__(WPB * 32) void moe_kernel(
    const float* __restrict__ cur, const float* __restrict__ nb,
    const int* __restrict__ cat,
    const float* __restrict__ b_local, const float* __restrict__ b_f1,
    const float* __restrict__ b_f2, const float* __restrict__ b_c1,
    const float* __restrict__ b_c2, const float* __restrict__ b_g1,
    const float* __restrict__ W_g2, const float* __restrict__ b_g2,
    const unsigned short* __restrict__ wpk, float* __restrict__ out) {
  __shared__ __align__(16) unsigned short smem[WAVE_LDS * WPB];
  int wave = threadIdx.x >> 5;
  int lane = threadIdx.x & 31;
  int tile = blockIdx.x * WPB + wave;
  if (tile >= TILES) return;                 // wave-uniform
  int row0 = tile * 16;
  unsigned short* lds = smem + wave * WAVE_LDS;
  unsigned short* XA = lds + XA_OFF;
  unsigned short* XH = lds + XH_OFF;
  float* wrow = (float*)(lds + WROW_OFF);
  float* XHf = (float*)(lds + XH_OFF);
  const int base8 = (lane >> 4) << 3;
  const int col = lane & 15;

  // ---- current_state -> XA[:,0:64] (bf16), perfectly coalesced ----
#pragma unroll 4
  for (int r = 0; r < 16; ++r) {
    int n = row0 + r; if (n > NCELLS - 1) n = NCELLS - 1;
    v2f c = *(const v2f*)(cur + (size_t)n * DDIM + lane * 2);
    XA[r * 136 + lane * 2 + 0] = f2bf(c.x);
    XA[r * 136 + lane * 2 + 1] = f2bf(c.y);
  }

  // ---- neighbor reduction: 3 masked sums + counts; act = total/26 ----
  // 131 MB single-use stream: non-temporal loads, deep unroll for HBM latency.
  for (int r = 0; r < 16; ++r) {
    int n = row0 + r; if (n > NCELLS - 1) n = NCELLS - 1;
    const float* nbp = nb + (size_t)n * (KNBR * DDIM) + lane * 2;
    const int* cp = cat + (size_t)n * KNBR;
    float a0x = 0, a0y = 0, a1x = 0, a1y = 0, a2x = 0, a2y = 0;
    int c0 = 0, c1 = 0, c2 = 0;
#pragma unroll 13
    for (int k = 0; k < KNBR; ++k) {
      int cc = cp[k];
      v2f v = __builtin_nontemporal_load((const v2f*)(nbp + k * DDIM));
      bool m0 = (cc == 0), m1 = (cc == 1), m2 = (cc == 2);
      a0x += m0 ? v.x : 0.0f; a0y += m0 ? v.y : 0.0f;
      a1x += m1 ? v.x : 0.0f; a1y += m1 ? v.y : 0.0f;
      a2x += m2 ? v.x : 0.0f; a2y += m2 ? v.y : 0.0f;
      c0 += m0; c1 += m1; c2 += m2;
    }
    float i0 = 1.0f / fmaxf((float)c0, 1.0f);
    float i1 = 1.0f / fmaxf((float)c1, 1.0f);
    float i2 = 1.0f / fmaxf((float)c2, 1.0f);
    unsigned short* m = lds + MEAN_OFF + r * 64 + lane * 2;
    m[0]    = f2bf(a0x * i0); m[1]    = f2bf(a0y * i0);
    m[1024] = f2bf(a1x * i1); m[1025] = f2bf(a1y * i1);
    m[2048] = f2bf(a2x * i2); m[2049] = f2bf(a2y * i2);
    XA[r * 136 + 64 + lane * 2 + 0] = f2bf((a0x + a1x + a2x) * (1.0f / 26.0f));
    XA[r * 136 + 64 + lane * 2 + 1] = f2bf((a0y + a1y + a2y) * (1.0f / 26.0f));
    if (lane == 0) {
      wrow[r * 8 + 4] = (float)c0;
      wrow[r * 8 + 5] = (float)c1;
      wrow[r * 8 + 6] = (float)c2;
    }
  }

  // ---- gating: [cur|act] @ W_g1 -> tanh -> @ W_g2 -> softmax ----
  {
    gemm128_fused<2>(XA, wpk + OFF_WG1, lane, [&](int t, v8f c) {
      float b = b_g1[t * 16 + col];
#pragma unroll
      for (int v = 0; v < 8; ++v) {
        XHf[(v + base8) * 32 + t * 16 + col] = fast_tanh(c[v] + b);
      }
    });
    if (lane < 16) {
      int rrow = lane;
      float g0 = b_g2[0], g1 = b_g2[1], g2v = b_g2[2];
#pragma unroll 4
      for (int j = 0; j < 32; ++j) {
        float h = XHf[rrow * 32 + j];
        g0  += h * W_g2[j * 3 + 0];
        g1  += h * W_g2[j * 3 + 1];
        g2v += h * W_g2[j * 3 + 2];
      }
      float mx = fmaxf(g0, fmaxf(g1, g2v));
      float e0 = __expf(g0 - mx), e1 = __expf(g1 - mx), e2 = __expf(g2v - mx);
      float inv = 1.0f / (e0 + e1 + e2);
      float w0 = e0 * inv, w1 = e1 * inv, w2 = e2 * inv;
      wrow[rrow * 8 + 0] = w0;
      wrow[rrow * 8 + 1] = w1;
      wrow[rrow * 8 + 2] = w2;
      int n = row0 + rrow;
      if (n < NCELLS) {
        float* wout = out + (size_t)NCELLS * DDIM + (size_t)n * 3;
        wout[0] = w0; wout[1] = w1; wout[2] = w2;
      }
    }
  }

  // combined accumulator: the ONLY expert state live across experts
  v8f comb[4];

  // ---- local expert: comb = w_l * tanh([cur|mean_l] @ W_local + b) ----
  {
    copy_mean(lds, 0, lane);
    float wl[8];
    load_gate_w(wrow, lane, 0, wl);
    gemm128_fused<4>(XA, wpk + OFF_WL, lane, [&](int t, v8f c) {
      float b = b_local[t * 16 + col];
#pragma unroll
      for (int v = 0; v < 8; ++v) { comb[t][v] = wl[v] * fast_tanh(c[v] + b); }
    });
  }

  // ---- functional expert (2-layer MLP): comb += w_f * out_f ----
  {
    copy_mean(lds, 1, lane);
    gemm128_fused<8>(XA, wpk + OFF_WF1, lane, [&](int t, v8f c) {
      float b = b_f1[t * 16 + col];
#pragma unroll
      for (int v = 0; v < 8; ++v) {
        XH[(v + base8) * 136 + t * 16 + col] = f2bf(fast_tanh(c[v] + b));
      }
    });
    float wf[8];
    load_gate_w(wrow, lane, 1, wf);
    gemm128_fused<4>(XH, wpk + OFF_WF2, lane, [&](int t, v8f c) {
      float b = b_f2[t * 16 + col];
#pragma unroll
      for (int v = 0; v < 8; ++v) { comb[t][v] += wf[v] * fast_tanh(c[v] + b); }
    });
  }

  // ---- distant expert: CNF, 3 Euler steps; comb += w_d * x_final ----
  {
    v8f xs[4];
    copy_mean(lds, 2, lane);
#pragma unroll
    for (int t = 0; t < 4; ++t) {
#pragma unroll
      for (int v = 0; v < 8; ++v) {
        int n = row0 + v + base8; if (n > NCELLS - 1) n = NCELLS - 1;
        xs[t][v] = cur[(size_t)n * DDIM + t * 16 + col];
      }
    }
    const float dt = 1.0f / 3.0f;
    for (int step = 0; step < 3; ++step) {
#pragma unroll
      for (int t = 0; t < 4; ++t) {
#pragma unroll
        for (int v = 0; v < 8; ++v) {
          XA[(v + base8) * 136 + t * 16 + col] = f2bf(xs[t][v]);
        }
      }
      gemm128_fused<8>(XA, wpk + OFF_WC1, lane, [&](int t, v8f c) {
        float b = b_c1[t * 16 + col];
#pragma unroll
        for (int v = 0; v < 8; ++v) {
          XH[(v + base8) * 136 + t * 16 + col] = f2bf(fast_tanh(c[v] + b));
        }
      });
      gemm128_fused<4>(XH, wpk + OFF_WC2, lane, [&](int t, v8f c) {
        float b = b_c2[t * 16 + col];
#pragma unroll
        for (int v = 0; v < 8; ++v) { xs[t][v] += dt * fast_tanh(c[v] + b); }
      });
    }
    float wd[8];
    load_gate_w(wrow, lane, 2, wd);
#pragma unroll
    for (int t = 0; t < 4; ++t) {
#pragma unroll
      for (int v = 0; v < 8; ++v) { comb[t][v] += wd[v] * xs[t][v]; }
    }
  }

  // ---- store combined ----
#pragma unroll
  for (int t = 0; t < 4; ++t) {
#pragma unroll
    for (int v = 0; v < 8; ++v) {
      int n = row0 + v + base8;
      if (n < NCELLS) { out[(size_t)n * DDIM + t * 16 + col] = comb[t][v]; }
    }
  }
}

// Pack f32 weight [128 x Nout] row-major -> bf16 B-fragment order:
// p = ((nt*4 + kb)*32 + lane)*16 + j ; value = W[kb*32 + (lane>=16?16:0) + j][nt*16 + (lane&15)]
__global__ void pack_w_kernel(const float* __restrict__ src, unsigned short* __restrict__ dst,
                              int Nout, int total) {
  int p = blockIdx.x * 256 + threadIdx.x;
  if (p >= total) return;
  int j = p & 15;
  int lane = (p >> 4) & 31;
  int rest = p >> 9;
  int kb = rest & 3;
  int nt = rest >> 2;
  int K = kb * 32 + ((lane >> 4) << 4) + j;
  int n = nt * 16 + (lane & 15);
  dst[p] = f2bf(src[K * Nout + n]);
}

extern "C" void kernel_launch(void* const* d_in, const int* in_sizes, int n_in,
                              void* d_out, int out_size, void* d_ws, size_t ws_size,
                              hipStream_t stream) {
  (void)in_sizes; (void)n_in; (void)out_size; (void)ws_size;
  unsigned short* wpk = (unsigned short*)d_ws;
  const struct { int idx; int off; int nout; } packs[6] = {
      {3, OFF_WL, 64},  {5, OFF_WF1, 128}, {7, OFF_WF2, 64},
      {9, OFF_WC1, 128}, {11, OFF_WC2, 64}, {13, OFF_WG1, 32}};
  for (int i = 0; i < 6; ++i) {
    int total = 128 * packs[i].nout;
    hipLaunchKernelGGL(pack_w_kernel, dim3((total + 255) / 256), dim3(256), 0, stream,
                       (const float*)d_in[packs[i].idx], wpk + packs[i].off,
                       packs[i].nout, total);
  }
  int blocks = (TILES + WPB - 1) / WPB;
  hipLaunchKernelGGL(moe_kernel, dim3(blocks), dim3(WPB * 32), 0, stream,
                     (const float*)d_in[0], (const float*)d_in[1], (const int*)d_in[2],
                     (const float*)d_in[4], (const float*)d_in[6], (const float*)d_in[8],
                     (const float*)d_in[10], (const float*)d_in[12], (const float*)d_in[14],
                     (const float*)d_in[15], (const float*)d_in[16],
                     wpk, (float*)d_out);
}
